// SpHourglass_71674414235858
// MI455X (gfx1250) — matrix operand order repaired and checked by reading, not compiled, not run on hardware
//
#include <hip/hip_runtime.h>

// ---------------------------------------------------------------------------
// Sparse hourglass UNet on gfx1250 (MI455X). Every conv is an implicit GEMM on
// v_wmma_f32_16x16x32_bf16 (wave32). Each wave computes a 16(voxel) x 32(Cout)
// tile: the A fragment (activations, the HBM-side traffic) is loaded once and
// reused for two WMMAs; B (weights) streams from L2. CIN and K are template
// parameters so the K-loop fully unrolls and loads pipeline ahead of the WMMAs.
// ---------------------------------------------------------------------------

typedef __attribute__((ext_vector_type(16))) __bf16 v16bf;
typedef __attribute__((ext_vector_type(8)))  float  v8f;

union Frag { v16bf v; uint4 q[2]; };

__device__ __forceinline__ float  bf2f(__bf16 x) { return (float)x; }
__device__ __forceinline__ __bf16 f2bf(float x)  { return (__bf16)x; }

// --------------------------- weight transform ------------------------------
// src f32 [T][Ci][Co]  ->  dst bf16 [T][Co][Ci]   (B-fragment friendly)
__global__ void wtrans_k(const float* __restrict__ w, __bf16* __restrict__ o,
                         int T, int Ci, int Co) {
  long i = (long)blockIdx.x * blockDim.x + threadIdx.x;
  long n = (long)T * Ci * Co;
  if (i >= n) return;
  int co = (int)(i % Co); long t = i / Co;
  int ci = (int)(t % Ci); int tp = (int)(t / Ci);
  o[((long)tp * Co + co) * Ci + ci] = f2bf(w[i]);
}

// --------------------------- mask 2x downsample ----------------------------
__global__ void maskdown_k(const unsigned char* __restrict__ mi,
                           unsigned char* __restrict__ mo,
                           int Bn, int cD, int cH, int cW) {
  long i = (long)blockIdx.x * blockDim.x + threadIdx.x;
  long n = (long)Bn * cD * cH * cW;
  if (i >= n) return;
  long t = i;
  int w = (int)(t % cW); t /= cW;
  int h = (int)(t % cH); t /= cH;
  int d = (int)(t % cD); int b = (int)(t / cD);
  int fD = 2 * cD, fH = 2 * cH, fW = 2 * cW;
  unsigned char acc = 0;
  for (int dz = 0; dz < 2; ++dz)
    for (int dy = 0; dy < 2; ++dy)
      for (int dx = 0; dx < 2; ++dx) {
        long fi = (((long)b * fD + 2 * d + dz) * fH + 2 * h + dy) * fW + 2 * w + dx;
        acc |= mi[fi];
      }
  mo[i] = acc ? 1 : 0;
}

// --------------------------- f32 -> bf16 with mask -------------------------
__global__ void maskmul_k(const float* __restrict__ x, __bf16* __restrict__ y,
                          int C, const unsigned char* __restrict__ mask, long n) {
  long i = (long)blockIdx.x * blockDim.x + threadIdx.x;
  if (i >= n) return;
  y[i] = f2bf(x[i] * (mask[i / C] ? 1.0f : 0.0f));
}

// --------------------------- BN(eval) + ReLU + mask ------------------------
__global__ void bnrelu_k(const __bf16* __restrict__ x, __bf16* __restrict__ y,
                         const float* __restrict__ g, const float* __restrict__ b,
                         const float* __restrict__ m, const float* __restrict__ v,
                         int chOff, int C, const unsigned char* __restrict__ mask,
                         long n) {
  long i = (long)blockIdx.x * blockDim.x + threadIdx.x;
  if (i >= n) return;
  int c = (int)(i % C); long vox = i / C;
  float xv = bf2f(x[i]);
  float yv = (xv - m[chOff + c]) * (g[chOff + c] * rsqrtf(v[chOff + c] + 1e-3f))
             + b[chOff + c];
  yv = fmaxf(yv, 0.0f) * (mask[vox] ? 1.0f : 0.0f);
  y[i] = f2bf(yv);
}

// --------------------------- implicit-GEMM conv ----------------------------
// One wave computes a 16(voxel) x 32(Cout) tile of  out = conv(A) [*mask][+resid].
// A is a (virtual) concat of A0 (C0 ch) and A1 (CIN-C0 ch); splits are 32-mult.
// Wt layout: [tap][Cout][CIN] bf16.
//   subm 3^3: K=3 s=1 p=1 ; down: K=2 s=2 p=0 ; 1x1 i-branch: K=1 s=1 p=0.
template <bool OUT_F32, int CIN, int K>
__global__ __launch_bounds__(256)
void conv_wmma_k(const __bf16* __restrict__ A0, int C0,
                 const __bf16* __restrict__ A1,
                 const __bf16* __restrict__ Wt,
                 const unsigned char* __restrict__ omask,
                 const __bf16* __restrict__ resid,
                 void* __restrict__ Out, int Cout,
                 int Bn, int inD, int inH, int inW,
                 int outD, int outH, int outW,
                 int stride, int pad) {
  const int lane = threadIdx.x & 31;
  const int wv   = threadIdx.x >> 5;
  const int ntiles = Cout >> 5;               // 32-wide N groups
  const long nVox = (long)Bn * outD * outH * outW;
  const long mtiles = (nVox + 15) >> 4;
  long tile = (long)blockIdx.x * 8 + wv;
  if (tile >= mtiles * (long)ntiles) return;
  const int  nb    = (int)(tile % ntiles);
  const long vbase = (tile / ntiles) * 16;

  // A row owned by this lane (lanes L and L+16 share a row, different K halves)
  const int  mrow = lane & 15;
  const long vox  = vbase + mrow;
  const bool rowvalid = vox < nVox;
  long t = rowvalid ? vox : 0;
  const int ow = (int)(t % outW); t /= outW;
  const int oh = (int)(t % outH); t /= outH;
  const int od = (int)(t % outD);
  const int ob = (int)(t / outD);

  const int koffA = (lane >> 4) * 8;   // A: 2x8 elems at +0/+16 (ISA 16-bit A layout)
  const int koffB = (lane >> 4) * 8;   // B: 2x8 elems at +0/+16 (matches A)
  // B rows for the two N-subtiles owned by this wave
  const __bf16* wrow0 = Wt + (long)(nb * 32 + mrow) * CIN;
  const __bf16* wrow1 = wrow0 + (long)16 * CIN;

  v8f acc0 = {}, acc1 = {};

  for (int kd = 0; kd < K; ++kd) {
    const int id = od * stride + kd - pad;
    const bool vd = rowvalid & ((unsigned)id < (unsigned)inD);
    for (int kh = 0; kh < K; ++kh) {
      const int ih = oh * stride + kh - pad;
      const bool vh = vd & ((unsigned)ih < (unsigned)inH);
      for (int kw = 0; kw < K; ++kw) {
        const int iw = ow * stride + kw - pad;
        const bool ok = vh & ((unsigned)iw < (unsigned)inW);
        const long ivox = (((long)ob * inD + id) * inH + ih) * inW + iw;
        const int  tap  = (kd * K + kh) * K + kw;
        const long wtapo = (long)tap * Cout * CIN;
#pragma unroll
        for (int cb = 0; cb < CIN; cb += 32) {
          Frag a, b0, b1;
          if (ok) {
            const __bf16* ap = (cb < C0) ? (A0 + ivox * C0 + cb)
                                         : (A1 + ivox * (CIN - C0) + (cb - C0));
            a.q[0] = *(const uint4*)(ap + koffA);
            a.q[1] = *(const uint4*)(ap + koffA + 16);
          } else {
            a.q[0] = make_uint4(0u, 0u, 0u, 0u);
            a.q[1] = make_uint4(0u, 0u, 0u, 0u);
          }
          const __bf16* bp0 = wrow0 + wtapo + cb + koffB;
          const __bf16* bp1 = wrow1 + wtapo + cb + koffB;
          b0.q[0] = *(const uint4*)(bp0);
          b0.q[1] = *(const uint4*)(bp0 + 16);
          b1.q[0] = *(const uint4*)(bp1);
          b1.q[1] = *(const uint4*)(bp1 + 16);
          acc0 = __builtin_amdgcn_wmma_f32_16x16x32_bf16(
              false, a.v, false, b0.v, (short)0, acc0, false, false);
          acc1 = __builtin_amdgcn_wmma_f32_16x16x32_bf16(
              false, a.v, false, b1.v, (short)0, acc1, false, false);
        }
      }
    }
  }

  // D layout: VGPR r -> M = r + 8*(lane>=16), N = lane%16
  const int mst = (lane >> 4) * 8;
#pragma unroll
  for (int nt = 0; nt < 2; ++nt) {
    const int ncol = nb * 32 + nt * 16 + (lane & 15);
    const v8f& acc = nt ? acc1 : acc0;
#pragma unroll
    for (int r = 0; r < 8; ++r) {
      const long svox = vbase + mst + r;
      if (svox < nVox) {
        float val = acc[r];
        if (resid) val += bf2f(resid[svox * Cout + ncol]);
        val *= omask[svox] ? 1.0f : 0.0f;
        if (OUT_F32) ((float*)Out)[svox * Cout + ncol] = val;
        else         ((__bf16*)Out)[svox * Cout + ncol] = f2bf(val);
      }
    }
  }
}

// --------------------------- 2x transpose-conv (k=2,s=2) -------------------
// out[2t+j, co] = sum_ci in[t, ci] * w[flip(j)][ci][co]; one wave handles a
// 16-coarse-voxel x 32-Cout tile for one parity j, scatter-stores to fine grid.
template <int CIN>
__global__ __launch_bounds__(256)
void upconv_wmma_k(const __bf16* __restrict__ A, const __bf16* __restrict__ Wt,
                   const unsigned char* __restrict__ omask,
                   __bf16* __restrict__ Out,
                   int Cout, int Bn, int cD, int cH, int cW) {
  const int lane = threadIdx.x & 31;
  const int wv   = threadIdx.x >> 5;
  const int ntiles = Cout >> 5;
  const long nCvox = (long)Bn * cD * cH * cW;
  const long mtiles = (nCvox + 15) >> 4;
  long tile = (long)blockIdx.x * 8 + wv;
  if (tile >= mtiles * (long)ntiles * 8) return;
  const int  j     = (int)(tile % 8);
  const int  nb    = (int)((tile / 8) % ntiles);
  const long vbase = (tile / (8 * (long)ntiles)) * 16;

  const int  mrow = lane & 15;
  const long vox  = vbase + mrow;
  const bool rowvalid = vox < nCvox;

  const int koffA = (lane >> 4) * 8;
  const int koffB = (lane >> 4) * 8;
  const int tap = 7 - j;  // conv_transpose uses spatially flipped taps
  const __bf16* wrow0 = Wt + ((long)tap * Cout + nb * 32 + mrow) * CIN;
  const __bf16* wrow1 = wrow0 + (long)16 * CIN;

  v8f acc0 = {}, acc1 = {};
#pragma unroll
  for (int cb = 0; cb < CIN; cb += 32) {
    Frag a, b0, b1;
    if (rowvalid) {
      const __bf16* ap = A + vox * CIN + cb;
      a.q[0] = *(const uint4*)(ap + koffA);
      a.q[1] = *(const uint4*)(ap + koffA + 16);
    } else {
      a.q[0] = make_uint4(0u, 0u, 0u, 0u);
      a.q[1] = make_uint4(0u, 0u, 0u, 0u);
    }
    const __bf16* bp0 = wrow0 + cb + koffB;
    const __bf16* bp1 = wrow1 + cb + koffB;
    b0.q[0] = *(const uint4*)(bp0);
    b0.q[1] = *(const uint4*)(bp0 + 16);
    b1.q[0] = *(const uint4*)(bp1);
    b1.q[1] = *(const uint4*)(bp1 + 16);
    acc0 = __builtin_amdgcn_wmma_f32_16x16x32_bf16(
        false, a.v, false, b0.v, (short)0, acc0, false, false);
    acc1 = __builtin_amdgcn_wmma_f32_16x16x32_bf16(
        false, a.v, false, b1.v, (short)0, acc1, false, false);
  }

  const int jd = j >> 2, jh = (j >> 1) & 1, jw = j & 1;
  const int fD = 2 * cD, fH = 2 * cH, fW = 2 * cW;
  const int mst = (lane >> 4) * 8;
#pragma unroll
  for (int nt = 0; nt < 2; ++nt) {
    const int ncol = nb * 32 + nt * 16 + (lane & 15);
    const v8f& acc = nt ? acc1 : acc0;
#pragma unroll
    for (int r = 0; r < 8; ++r) {
      const long svox = vbase + mst + r;
      if (svox < nCvox) {
        long t = svox;
        const int cw = (int)(t % cW); t /= cW;
        const int ch = (int)(t % cH); t /= cH;
        const int cd = (int)(t % cD);
        const int ob = (int)(t / cD);
        const long fvox = (((long)ob * fD + (2 * cd + jd)) * fH + (2 * ch + jh)) * fW
                          + (2 * cw + jw);
        Out[fvox * Cout + ncol] = f2bf(acc[r] * (omask[fvox] ? 1.0f : 0.0f));
      }
    }
  }
}

// ===========================================================================
extern "C" void kernel_launch(void* const* d_in, const int* in_sizes, int n_in,
                              void* d_out, int out_size, void* d_ws, size_t ws_size,
                              hipStream_t stream) {
  (void)in_sizes; (void)n_in; (void)out_size; (void)ws_size;
  const int Bn = 2;
  const int D1 = 96, H1 = 96, W1 = 48;
  const int D2 = 48, H2 = 48, W2 = 24;
  const int D3 = 24, H3 = 24, W3 = 12;
  const long vox1 = (long)Bn * D1 * H1 * W1;   // 884736
  const long vox2 = (long)Bn * D2 * H2 * W2;   // 110592
  const long vox3 = (long)Bn * D3 * H3 * W3;   // 13824

  // d_in: pytree flattened with JAX-sorted dict keys:
  // 0 features, 1 mask, then params alphabetically; bn dicts -> b,g,m,v.
  const float* features = (const float*)d_in[0];
  const unsigned char* m1 = (const unsigned char*)d_in[1];  // numpy bool
  auto F = [&](int i) { return (const float*)d_in[i]; };

  // ---------------- workspace layout (bf16 activations, ~400MB) ----------
  char* ws = (char*)d_ws;
  size_t off = 0;
  auto alloc = [&](size_t bytes) {
    off = (off + 255) & ~(size_t)255;
    size_t o = off; off += bytes; return o;
  };
  auto BF = [&](size_t o) { return (__bf16*)(ws + o); };

  struct WD { int idx, T, Ci, Co; size_t off; };
  WD wd[16] = {
      {10, 27, 32, 32, 0},   // 0  l1_block.w1
      {11, 27, 32, 32, 0},   // 1  l1_block.w2
      {16, 8, 32, 64, 0},    // 2  l1_down_w
      {25, 27, 64, 32, 0},   // 3  l1_tail.w1
      {26, 27, 32, 32, 0},   // 4  l1_tail.w2
      {27, 1, 64, 32, 0},    // 5  l1_tail.wi
      {32, 8, 64, 32, 0},    // 6  l1_up_w
      {41, 27, 64, 64, 0},   // 7  l2_block.w1
      {42, 27, 64, 64, 0},   // 8  l2_block.w2
      {47, 8, 64, 128, 0},   // 9  l2_down_w
      {56, 27, 128, 64, 0},  // 10 l2_tail.w1
      {57, 27, 64, 64, 0},   // 11 l2_tail.w2
      {58, 1, 128, 64, 0},   // 12 l2_tail.wi
      {63, 8, 128, 64, 0},   // 13 l2_up_w
      {72, 27, 128, 128, 0}, // 14 l3_block.w1
      {73, 27, 128, 128, 0}  // 15 l3_block.w2
  };
  for (int i = 0; i < 16; ++i)
    wd[i].off = alloc((size_t)wd[i].T * wd[i].Ci * wd[i].Co * 2);

  size_t oM2 = alloc(vox2), oM3 = alloc(vox3);
  size_t oX1a = alloc(vox1 * 32 * 2), oT1 = alloc(vox1 * 32 * 2),
         oH1 = alloc(vox1 * 32 * 2),  oX1 = alloc(vox1 * 32 * 2),
         oUp1 = alloc(vox1 * 32 * 2);
  size_t oX2 = alloc(vox2 * 64 * 2), oT2a = alloc(vox2 * 64 * 2),
         oT2b = alloc(vox2 * 64 * 2), oH2 = alloc(vox2 * 64 * 2),
         oUp2 = alloc(vox2 * 64 * 2), oX2t = alloc(vox2 * 64 * 2),
         oIB2 = alloc(vox2 * 64 * 2);
  size_t oX3 = alloc(vox3 * 128 * 2), oT3 = alloc(vox3 * 128 * 2),
         oH3 = alloc(vox3 * 128 * 2);

  unsigned char* m2 = (unsigned char*)(ws + oM2);
  unsigned char* m3 = (unsigned char*)(ws + oM3);

  // ---------------- prep: weights -> bf16 [tap][Co][Ci], pooled masks ----
  for (int i = 0; i < 16; ++i) {
    long n = (long)wd[i].T * wd[i].Ci * wd[i].Co;
    wtrans_k<<<(int)((n + 255) / 256), 256, 0, stream>>>(
        F(wd[i].idx), BF(wd[i].off), wd[i].T, wd[i].Ci, wd[i].Co);
  }
  maskdown_k<<<(int)((vox2 + 255) / 256), 256, 0, stream>>>(m1, m2, Bn, D2, H2, W2);
  maskdown_k<<<(int)((vox3 + 255) / 256), 256, 0, stream>>>(m2, m3, Bn, D3, H3, W3);
  {
    long n = vox1 * 32;
    maskmul_k<<<(int)((n + 255) / 256), 256, 0, stream>>>(features, BF(oX1a), 32, m1, n);
  }

  // ---------------- helpers ---------------------------------------------
  auto bnr = [&](size_t xo, size_t yo, int base, int C, int chOff,
                 const unsigned char* mk, long nvox) {
    long n = nvox * C;
    bnrelu_k<<<(int)((n + 255) / 256), 256, 0, stream>>>(
        BF(xo), BF(yo), F(base + 1), F(base + 0), F(base + 2), F(base + 3),
        chOff, C, mk, n);
  };

  auto conv = [&](size_t a0, int C0, size_t a1, int Cc1, int wi,
                  const unsigned char* om, size_t ro, bool hasResid,
                  void* out, bool outf32, int Cout,
                  int iD, int iH, int iW, int oD, int oH, int oW,
                  int K, int stride, int pad) {
    long nVox = (long)Bn * oD * oH * oW;
    long waves = ((nVox + 15) / 16) * (Cout / 32);
    int blocks = (int)((waves + 7) / 8);
    const __bf16* A0p = BF(a0);
    const __bf16* A1p = Cc1 ? BF(a1) : nullptr;
    const __bf16* Rp = hasResid ? BF(ro) : nullptr;
    const __bf16* Wp = BF(wd[wi].off);
    int Cin = C0 + Cc1;
#define LC(CIN_, K_)                                                         \
    do {                                                                     \
      if (outf32)                                                            \
        conv_wmma_k<true, CIN_, K_><<<blocks, 256, 0, stream>>>(             \
            A0p, C0, A1p, Wp, om, Rp, out, Cout, Bn, iD, iH, iW,             \
            oD, oH, oW, stride, pad);                                        \
      else                                                                   \
        conv_wmma_k<false, CIN_, K_><<<blocks, 256, 0, stream>>>(            \
            A0p, C0, A1p, Wp, om, Rp, out, Cout, Bn, iD, iH, iW,             \
            oD, oH, oW, stride, pad);                                        \
    } while (0)
    if (Cin == 32) {
      if (K == 3) LC(32, 3); else if (K == 2) LC(32, 2); else LC(32, 1);
    } else if (Cin == 64) {
      if (K == 3) LC(64, 3); else if (K == 2) LC(64, 2); else LC(64, 1);
    } else {
      if (K == 3) LC(128, 3); else if (K == 2) LC(128, 2); else LC(128, 1);
    }
#undef LC
  };

  auto upconv = [&](size_t a, int wi, const unsigned char* om, size_t outo,
                    int Cin, int Cout, int cD, int cH, int cW) {
    long nCvox = (long)Bn * cD * cH * cW;
    long waves = ((nCvox + 15) / 16) * (Cout / 32) * 8;
    int blocks = (int)((waves + 7) / 8);
    if (Cin == 64)
      upconv_wmma_k<64><<<blocks, 256, 0, stream>>>(
          BF(a), BF(wd[wi].off), om, BF(outo), Cout, Bn, cD, cH, cW);
    else
      upconv_wmma_k<128><<<blocks, 256, 0, stream>>>(
          BF(a), BF(wd[wi].off), om, BF(outo), Cout, Bn, cD, cH, cW);
  };

  // ---------------- level 1 encode ---------------------------------------
  bnr(oX1a, oT1, 2, 32, 0, m1, vox1);                                      // bn1
  conv(oT1, 32, 0, 0, 0, m1, 0, false, BF(oH1), false, 32, D1, H1, W1, D1, H1, W1, 3, 1, 1);
  bnr(oH1, oT1, 6, 32, 0, m1, vox1);                                       // bn2
  conv(oT1, 32, 0, 0, 1, m1, oX1a, true, BF(oX1), false, 32, D1, H1, W1, D1, H1, W1, 3, 1, 1);
  bnr(oX1, oT1, 12, 32, 0, m1, vox1);                                      // l1_down_bn
  conv(oT1, 32, 0, 0, 2, m2, 0, false, BF(oX2), false, 64, D1, H1, W1, D2, H2, W2, 2, 2, 0);

  // ---------------- level 2 encode ---------------------------------------
  bnr(oX2, oT2a, 33, 64, 0, m2, vox2);
  conv(oT2a, 64, 0, 0, 7, m2, 0, false, BF(oH2), false, 64, D2, H2, W2, D2, H2, W2, 3, 1, 1);
  bnr(oH2, oT2a, 37, 64, 0, m2, vox2);
  conv(oT2a, 64, 0, 0, 8, m2, oX2, true, BF(oX2), false, 64, D2, H2, W2, D2, H2, W2, 3, 1, 1);
  bnr(oX2, oT2a, 43, 64, 0, m2, vox2);                                     // l2_down_bn
  conv(oT2a, 64, 0, 0, 9, m3, 0, false, BF(oX3), false, 128, D2, H2, W2, D3, H3, W3, 2, 2, 0);

  // ---------------- level 3 (bottom) -------------------------------------
  bnr(oX3, oT3, 64, 128, 0, m3, vox3);
  conv(oT3, 128, 0, 0, 14, m3, 0, false, BF(oH3), false, 128, D3, H3, W3, D3, H3, W3, 3, 1, 1);
  bnr(oH3, oT3, 68, 128, 0, m3, vox3);
  conv(oT3, 128, 0, 0, 15, m3, oX3, true, BF(oX3), false, 128, D3, H3, W3, D3, H3, W3, 3, 1, 1);

  // ---------------- decode to level 2 ------------------------------------
  bnr(oX3, oT3, 59, 128, 0, m3, vox3);                                     // l2_up_bn
  upconv(oT3, 13, m2, oUp2, 128, 64, D3, H3, W3);
  // l2_tail on concat(x2, up2): i-branch 1x1, then two subm convs
  conv(oX2, 64, oUp2, 64, 12, m2, 0, false, BF(oIB2), false, 64, D2, H2, W2, D2, H2, W2, 1, 1, 0);
  bnr(oX2, oT2a, 48, 64, 0, m2, vox2);                                     // bn1[0:64]
  bnr(oUp2, oT2b, 48, 64, 64, m2, vox2);                                   // bn1[64:128]
  conv(oT2a, 64, oT2b, 64, 10, m2, 0, false, BF(oH2), false, 64, D2, H2, W2, D2, H2, W2, 3, 1, 1);
  bnr(oH2, oT2a, 52, 64, 0, m2, vox2);                                     // bn2
  conv(oT2a, 64, 0, 0, 11, m2, oIB2, true, BF(oX2t), false, 64, D2, H2, W2, D2, H2, W2, 3, 1, 1);

  // ---------------- decode to level 1 ------------------------------------
  bnr(oX2t, oT2a, 28, 64, 0, m2, vox2);                                    // l1_up_bn
  upconv(oT2a, 6, m1, oUp1, 64, 32, D2, H2, W2);
  // l1_tail on concat(x1, up1): i-branch into oX1a (free), output f32 -> d_out
  conv(oX1, 32, oUp1, 32, 5, m1, 0, false, BF(oX1a), false, 32, D1, H1, W1, D1, H1, W1, 1, 1, 0);
  bnr(oX1, oT1, 17, 32, 0, m1, vox1);                                      // bn1[0:32]
  bnr(oUp1, oH1, 17, 32, 32, m1, vox1);                                    // bn1[32:64]
  conv(oT1, 32, oH1, 32, 3, m1, 0, false, BF(oUp1), false, 32, D1, H1, W1, D1, H1, W1, 3, 1, 1);
  bnr(oUp1, oT1, 21, 32, 0, m1, vox1);                                     // bn2
  conv(oT1, 32, 0, 0, 4, m1, oX1a, true, d_out, true, 32, D1, H1, W1, D1, H1, W1, 3, 1, 1);
}
